// Net_60129542144705
// MI455X (gfx1250) — matrix operand order repaired and checked by reading, not compiled, not run on hardware
//
#include <hip/hip_runtime.h>
#include <hip/hip_bf16.h>
#include <math.h>

#define N_NODES   50000
#define N_EDGES   800000
#define IN_FEAT   128
#define HIDDEN    16
#define N_CLASSES 40

typedef __attribute__((ext_vector_type(16))) __bf16 v16bf;
typedef __attribute__((ext_vector_type(8)))  float  v8f;

__device__ __forceinline__ v8f wmma_bf16(v16bf a, v16bf b, v8f c) {
  // D = A(16x32 bf16) x B(32x16 bf16) + C(16x16 f32)
  return __builtin_amdgcn_wmma_f32_16x16x32_bf16(false, a, false, b, (short)0, c, false, false);
}

// Scatter-max straight to L2 atomic units, no return value (STOREcnt path).
__device__ __forceinline__ void atomic_max_f32(float* addr, float val) {
  asm volatile("global_atomic_max_num_f32 %0, %1, off scope:SCOPE_DEV"
               :: "v"(addr), "v"(val)
               : "memory");
}

__global__ void fill_kernel(float* __restrict__ p, int n, float v) {
  int i = blockIdx.x * blockDim.x + threadIdx.x;
  if (i < n) p[i] = v;
}

// ---------------- Layer 1: per 16-edge tile ----------------
// A = [x_i | x_j - x_i] (16x256 bf16), B = W1a (256x16), then
// relu(+b1a) -> LDS -> A2 (16x32, K>=16 zero), B2 = W1b (padded 32x16).
__global__ __launch_bounds__(256) void edgeconv1_kernel(
    const float* __restrict__ x, const int* __restrict__ ei,
    const float* __restrict__ W1a, const float* __restrict__ b1a,
    const float* __restrict__ W1b, const float* __restrict__ b1b,
    float* __restrict__ h1)
{
  __shared__ float lds[8 * 16 * 16];
  const int lane  = threadIdx.x & 31;
  const int m     = lane & 15;   // A-row / B-col / C-col for this lane
  const int khalf = lane >> 4;
  float* lp = lds + (threadIdx.x >> 5) * 256;

  // ---- preload weight fragments (amortized over the tile loop) ----
  v16bf bW1a[8];
#pragma unroll
  for (int kk = 0; kk < 8; ++kk) {
#pragma unroll
    for (int e = 0; e < 16; ++e) {
      int k = kk * 32 + khalf * 16 + e;            // B layout: K = khalf*16+e
      bW1a[kk][e] = (__bf16)W1a[k * HIDDEN + m];
    }
  }
  v16bf bW1b;
#pragma unroll
  for (int e = 0; e < 16; ++e) {
    int k = (e < 8) ? (khalf * 8 + e) : (16 + khalf * 8 + (e - 8));
    float wv = (k < HIDDEN) ? W1b[k * HIDDEN + m] : 0.0f;  // pad K 16->32
    bW1b[e] = (__bf16)wv;
  }
  const float bias1a = b1a[m];
  const float bias1b = b1b[m];

  const int nwaves = (gridDim.x * blockDim.x) >> 5;
  const int wid    = (blockIdx.x * blockDim.x + threadIdx.x) >> 5;
  const int ntiles = N_EDGES / 16;

  for (int t = wid; t < ntiles; t += nwaves) {
    const int eb   = t * 16;
    const int sidx = ei[eb + m];                 // src of edge (eb+m)
    const int didx = ei[N_EDGES + eb + m];       // dst of edge (eb+m)
    const float* xi = x + (size_t)didx * IN_FEAT;
    const float* xj = x + (size_t)sidx * IN_FEAT;

    int drow[8];                                 // dst per C-row, pre-shuffled
#pragma unroll
    for (int r = 0; r < 8; ++r) drow[r] = __shfl(didx, r + khalf * 8);

    v8f acc = {0.f,0.f,0.f,0.f,0.f,0.f,0.f,0.f};
#pragma unroll
    for (int kk = 0; kk < 8; ++kk) {
      v16bf a;
#pragma unroll
      for (int e = 0; e < 16; ++e) {
        int K = khalf * 16 + e;
        int c = kk * 32 + K;                     // constant-folds: c<128 <=> kk<4
        float v = (c < IN_FEAT) ? xi[c] : (xj[c - IN_FEAT] - xi[c - IN_FEAT]);
        a[e] = (__bf16)v;
      }
      acc = wmma_bf16(a, bW1a[kk], acc);
    }

    // relu(acc + b1a) -> LDS (16x16 row-major, this wave's slot)
#pragma unroll
    for (int r = 0; r < 8; ++r) {
      int mrow = r + khalf * 8;
      lp[mrow * 16 + m] = fmaxf(acc[r] + bias1a, 0.0f);
    }
    __builtin_amdgcn_wave_barrier();

    v16bf a2;
#pragma unroll
    for (int e = 0; e < 16; ++e) {
      float v = (e < 8) ? lp[m * 16 + khalf * 8 + e] : 0.0f;  // K>=16 padded 0
      a2[e] = (__bf16)v;
    }
    __builtin_amdgcn_wave_barrier();

    v8f z = {0.f,0.f,0.f,0.f,0.f,0.f,0.f,0.f};
    v8f acc2 = wmma_bf16(a2, bW1b, z);

#pragma unroll
    for (int r = 0; r < 8; ++r)
      atomic_max_f32(h1 + (size_t)drow[r] * HIDDEN + m, acc2[r] + bias1b);
  }
}

// ---------------- Layer 2: per 16-edge tile ----------------
// gather applies max(h,0) (folds segment_max -inf fill + relu).
// A = [h_i | h_j - h_i] (16x32) -> 1 WMMA per 16-col tile of W2a (40->48),
// relu(+b2a) -> LDS (16x48) -> A2 (2 K-blocks, K 48..63 zero) x W2b (64x48 padded).
__global__ __launch_bounds__(256) void edgeconv2_kernel(
    const float* __restrict__ h1, const int* __restrict__ ei,
    const float* __restrict__ W2a, const float* __restrict__ b2a,
    const float* __restrict__ W2b, const float* __restrict__ b2b,
    float* __restrict__ out)
{
  __shared__ float lds[8 * 16 * 48];
  const int lane  = threadIdx.x & 31;
  const int m     = lane & 15;
  const int khalf = lane >> 4;
  float* lp = lds + (threadIdx.x >> 5) * (16 * 48);

  v16bf bA[3];
#pragma unroll
  for (int tn = 0; tn < 3; ++tn) {
    int col = tn * 16 + m;
#pragma unroll
    for (int e = 0; e < 16; ++e) {
      int k = (e < 8) ? (khalf * 8 + e) : (16 + khalf * 8 + (e - 8));  // 0..31 (full K)
      float wv = (col < N_CLASSES) ? W2a[k * N_CLASSES + col] : 0.0f;
      bA[tn][e] = (__bf16)wv;
    }
  }
  v16bf bB[2][3];
#pragma unroll
  for (int kb = 0; kb < 2; ++kb)
#pragma unroll
    for (int tn = 0; tn < 3; ++tn) {
      int col = tn * 16 + m;
#pragma unroll
      for (int e = 0; e < 16; ++e) {
        int k = kb * 32 + khalf * 16 + e;        // pad K 40->64
        float wv = (k < N_CLASSES && col < N_CLASSES) ? W2b[k * N_CLASSES + col] : 0.0f;
        bB[kb][tn][e] = (__bf16)wv;
      }
    }
  float biasA[3], biasB[3];
#pragma unroll
  for (int tn = 0; tn < 3; ++tn) {
    int col = tn * 16 + m;
    biasA[tn] = (col < N_CLASSES) ? b2a[col] : 0.0f;
    biasB[tn] = (col < N_CLASSES) ? b2b[col] : 0.0f;
  }

  const int nwaves = (gridDim.x * blockDim.x) >> 5;
  const int wid    = (blockIdx.x * blockDim.x + threadIdx.x) >> 5;
  const int ntiles = N_EDGES / 16;

  for (int t = wid; t < ntiles; t += nwaves) {
    const int eb   = t * 16;
    const int sidx = ei[eb + m];
    const int didx = ei[N_EDGES + eb + m];

    int drow[8];
#pragma unroll
    for (int r = 0; r < 8; ++r) drow[r] = __shfl(didx, r + khalf * 8);

    float vi[8], vj[8];
#pragma unroll
    for (int i = 0; i < 8; ++i) {
      vi[i] = fmaxf(h1[(size_t)didx * HIDDEN + khalf * 8 + i], 0.0f);
      vj[i] = fmaxf(h1[(size_t)sidx * HIDDEN + khalf * 8 + i], 0.0f);
    }
    v16bf a;
#pragma unroll
    for (int e = 0; e < 16; ++e) {
      float v = (e < 8) ? vi[e] : (vj[e - 8] - vi[e - 8]);
      a[e] = (__bf16)v;
    }

    v8f z = {0.f,0.f,0.f,0.f,0.f,0.f,0.f,0.f};
    v8f c1[3];
#pragma unroll
    for (int tn = 0; tn < 3; ++tn) c1[tn] = wmma_bf16(a, bA[tn], z);

#pragma unroll
    for (int tn = 0; tn < 3; ++tn)
#pragma unroll
      for (int r = 0; r < 8; ++r) {
        int mrow = r + khalf * 8;
        lp[mrow * 48 + tn * 16 + m] = fmaxf(c1[tn][r] + biasA[tn], 0.0f);
      }
    __builtin_amdgcn_wave_barrier();

    v16bf a2[2];
#pragma unroll
    for (int kb = 0; kb < 2; ++kb)
#pragma unroll
      for (int e = 0; e < 16; ++e) {
        int K = kb * 32 + khalf * 16 + e;
        float v = (K < 48) ? lp[m * 48 + K] : 0.0f;   // zero only for kb=1, khalf=1 (K 48..63)
        a2[kb][e] = (__bf16)v;
      }
    __builtin_amdgcn_wave_barrier();

    v8f c2[3] = {z, z, z};
#pragma unroll
    for (int kb = 0; kb < 2; ++kb)
#pragma unroll
      for (int tn = 0; tn < 3; ++tn)
        c2[tn] = wmma_bf16(a2[kb], bB[kb][tn], c2[tn]);

#pragma unroll
    for (int tn = 0; tn < 3; ++tn) {
      int col = tn * 16 + m;
      if (col < N_CLASSES) {
#pragma unroll
        for (int r = 0; r < 8; ++r)
          atomic_max_f32(out + (size_t)drow[r] * N_CLASSES + col, c2[tn][r] + biasB[tn]);
      }
    }
  }
}

// In-place: replace -inf (nodes with no in-edges) with 0, then log_softmax per row.
__global__ void logsoftmax_kernel(float* __restrict__ out) {
  int i = blockIdx.x * blockDim.x + threadIdx.x;
  if (i >= N_NODES) return;
  float v[N_CLASSES];
  float mx = -3.4e38f;
#pragma unroll
  for (int j = 0; j < N_CLASSES; ++j) {
    float t = out[i * N_CLASSES + j];
    if (t == -INFINITY) t = 0.0f;
    v[j] = t;
    mx = fmaxf(mx, t);
  }
  float s = 0.0f;
#pragma unroll
  for (int j = 0; j < N_CLASSES; ++j) s += __expf(v[j] - mx);
  float lse = mx + __logf(s);
#pragma unroll
  for (int j = 0; j < N_CLASSES; ++j) out[i * N_CLASSES + j] = v[j] - lse;
}

extern "C" void kernel_launch(void* const* d_in, const int* in_sizes, int n_in,
                              void* d_out, int out_size, void* d_ws, size_t ws_size,
                              hipStream_t stream) {
  (void)in_sizes; (void)n_in; (void)out_size; (void)ws_size;
  const float* x   = (const float*)d_in[0];
  const int*   ei  = (const int*)d_in[1];
  const float* W1a = (const float*)d_in[2];
  const float* b1a = (const float*)d_in[3];
  const float* W1b = (const float*)d_in[4];
  const float* b1b = (const float*)d_in[5];
  const float* W2a = (const float*)d_in[6];
  const float* b2a = (const float*)d_in[7];
  const float* W2b = (const float*)d_in[8];
  const float* b2b = (const float*)d_in[9];
  float* out = (float*)d_out;
  float* h1  = (float*)d_ws;          // 50000*16 f32 = 3.2 MB scratch

  const int nh1  = N_NODES * HIDDEN;
  const int nout = N_NODES * N_CLASSES;
  fill_kernel<<<(nh1  + 255) / 256, 256, 0, stream>>>(h1,  nh1,  -INFINITY);
  fill_kernel<<<(nout + 255) / 256, 256, 0, stream>>>(out, nout, -INFINITY);

  edgeconv1_kernel<<<1024, 256, 0, stream>>>(x, ei, W1a, b1a, W1b, b1b, h1);
  edgeconv2_kernel<<<1024, 256, 0, stream>>>(h1, ei, W2a, b2a, W2b, b2b, out);
  logsoftmax_kernel<<<(N_NODES + 255) / 256, 256, 0, stream>>>(out);
}
